// GAT_2516850835924
// MI455X (gfx1250) — matrix operand order repaired and checked by reading, not compiled, not run on hardware
//
#include <hip/hip_runtime.h>
#include <hip/hip_bf16.h>

typedef __attribute__((ext_vector_type(16))) __bf16 v16bf;
typedef __attribute__((ext_vector_type(8)))  __bf16 bf16x8;
typedef __attribute__((ext_vector_type(8)))  float  v8f;

#define HEADS 4
#define F     256   // feature width both layers (HEADS*64)
#define CDIM  64

// ---- monotonic float<->u32 ordering for atomic max on floats ----
__device__ __forceinline__ unsigned f2o(float f) {
  unsigned u = __float_as_uint(f);
  return (u & 0x80000000u) ? ~u : (u | 0x80000000u);
}
__device__ __forceinline__ float o2f(unsigned u) {
  u = (u & 0x80000000u) ? (u ^ 0x80000000u) : ~u;
  return __uint_as_float(u);
}

// ---- elementwise f32 -> bf16 ----
__global__ void k_cvt_bf16(const float* __restrict__ in, __bf16* __restrict__ out, int n) {
  int i = blockIdx.x * blockDim.x + threadIdx.x;
  if (i < n) out[i] = (__bf16)in[i];
}

// ---- W [256,256] f32 row-major -> Wt [256,256] bf16 (transposed) ----
__global__ void k_transpose_cvt(const float* __restrict__ W, __bf16* __restrict__ Wt) {
  int i = blockIdx.x * blockDim.x + threadIdx.x;   // 65536 elems
  int j = i >> 8, k = i & 255;                     // Wt[j][k] = W[k][j]
  Wt[i] = (__bf16)W[k * 256 + j];
}

// ---- WMMA bf16 GEMM fused with per-head attention dot products ----
// 1 wave/block; wave computes 32(M) x 64(N); blockIdx.y == head (64-col block).
// Software-pipelined: next K-step fragments are loaded before current is consumed.
// Epilogue writes C and a_src/a_dst via 16-lane shuffle reduction (no extra pass).
__global__ __launch_bounds__(32) void k_gemm_att(
    const __bf16* __restrict__ A, const __bf16* __restrict__ Bt,
    const float* __restrict__ attS, const float* __restrict__ attD,
    float* __restrict__ C, float* __restrict__ aS, float* __restrict__ aD, int M)
{
  const int lane = threadIdx.x;
  const int m    = lane & 15;
  const int half = lane >> 4;
  const int hd   = blockIdx.y;          // head index == 64-column block
  const int nBase = hd * 64;

  int row0 = blockIdx.x * 32 + m;
  int row1 = row0 + 16;
  const int r0c = row0 < M ? row0 : M - 1;   // clamp loads; stores guarded
  const int r1c = row1 < M ? row1 : M - 1;
  const __bf16* arow0 = A + (size_t)r0c * F;
  const __bf16* arow1 = A + (size_t)r1c * F;

  v8f acc[2][4] = {};

  // current fragments (A 16x32 layout: lane=M, elems 0..7 -> K=8*half+e,
  // elems 8..15 -> K=16+8*half+e; B 32x16: lane=col, same K mapping as A)
  bf16x8 a0l, a0h, a1l, a1h, bl[4], bh[4];
  a0l = *(const bf16x8*)(arow0 + 8 * half);
  a0h = *(const bf16x8*)(arow0 + 16 + 8 * half);
  a1l = *(const bf16x8*)(arow1 + 8 * half);
  a1h = *(const bf16x8*)(arow1 + 16 + 8 * half);
  #pragma unroll
  for (int s = 0; s < 4; ++s) {
    const __bf16* bcol = Bt + (size_t)(nBase + s * 16 + m) * F + 8 * half;
    bl[s] = *(const bf16x8*)(bcol);
    bh[s] = *(const bf16x8*)(bcol + 16);
  }

  #pragma unroll
  for (int k0 = 0; k0 < F; k0 += 32) {
    bf16x8 na0l = {}, na0h = {}, na1l = {}, na1h = {}, nbl[4] = {}, nbh[4] = {};
    if (k0 + 32 < F) {                   // prefetch next K-step
      const int kn = k0 + 32;
      na0l = *(const bf16x8*)(arow0 + kn + 8 * half);
      na0h = *(const bf16x8*)(arow0 + kn + 16 + 8 * half);
      na1l = *(const bf16x8*)(arow1 + kn + 8 * half);
      na1h = *(const bf16x8*)(arow1 + kn + 16 + 8 * half);
      #pragma unroll
      for (int s = 0; s < 4; ++s) {
        const __bf16* bcol = Bt + (size_t)(nBase + s * 16 + m) * F + kn + 8 * half;
        nbl[s] = *(const bf16x8*)(bcol);
        nbh[s] = *(const bf16x8*)(bcol + 16);
      }
    }
    v16bf a0, a1;
    #pragma unroll
    for (int i = 0; i < 8; ++i) {
      a0[i] = a0l[i]; a0[8 + i] = a0h[i];
      a1[i] = a1l[i]; a1[8 + i] = a1h[i];
    }
    #pragma unroll
    for (int s = 0; s < 4; ++s) {
      v16bf b;
      #pragma unroll
      for (int i = 0; i < 8; ++i) { b[i] = bl[s][i]; b[8 + i] = bh[s][i]; }
      acc[0][s] = __builtin_amdgcn_wmma_f32_16x16x32_bf16(
          false, a0, false, b, (short)0, acc[0][s], false, false);
      acc[1][s] = __builtin_amdgcn_wmma_f32_16x16x32_bf16(
          false, a1, false, b, (short)0, acc[1][s], false, false);
    }
    a0l = na0l; a0h = na0h; a1l = na1l; a1h = na1h;
    #pragma unroll
    for (int s = 0; s < 4; ++s) { bl[s] = nbl[s]; bh[s] = nbh[s]; }
  }

  // ---- epilogue: store C, fused a_src/a_dst for this head ----
  float asv[4], adv[4];
  #pragma unroll
  for (int s = 0; s < 4; ++s) {
    asv[s] = attS[nBase + s * 16 + m];
    adv[s] = attD[nBase + s * 16 + m];
  }
  #pragma unroll
  for (int t = 0; t < 2; ++t) {
    const int rb = blockIdx.x * 32 + t * 16 + half * 8;
    #pragma unroll
    for (int r = 0; r < 8; ++r) {
      const int rr = rb + r;
      float ps = 0.f, pd = 0.f;
      #pragma unroll
      for (int s = 0; s < 4; ++s) {
        float v = acc[t][s][r];
        if (rr < M) C[(size_t)rr * F + nBase + s * 16 + m] = v;
        ps += v * asv[s];
        pd += v * adv[s];
      }
      // reduce over the 16 lanes (m) that cover this head's 64 columns
      #pragma unroll
      for (int mask = 8; mask >= 1; mask >>= 1) {
        ps += __shfl_xor(ps, mask);
        pd += __shfl_xor(pd, mask);
      }
      if (m == 0 && rr < M) {
        aS[rr * HEADS + hd] = ps;
        aD[rr * HEADS + hd] = pd;
      }
    }
  }
}

__global__ void k_init_stats(unsigned* __restrict__ nmax, float* __restrict__ den, int n) {
  int i = blockIdx.x * blockDim.x + threadIdx.x;
  if (i < n) { nmax[i] = f2o(-3.0e38f); den[i] = 0.f; }
}

__global__ void k_zero(float4* __restrict__ p, int n4) {
  int i = blockIdx.x * blockDim.x + threadIdx.x;
  if (i < n4) p[i] = make_float4(0.f, 0.f, 0.f, 0.f);
}

// ---- per-edge leaky-relu score + segment max (u32 atomic on ordered bits) ----
__global__ void k_edge_max(const long long* __restrict__ src, const long long* __restrict__ dst,
                           const float* __restrict__ a_s, const float* __restrict__ a_d,
                           float* __restrict__ ebuf, unsigned* __restrict__ nmax,
                           int E, int Etot) {
  int i = blockIdx.x * blockDim.x + threadIdx.x;   // over Etot*HEADS
  if (i >= Etot * HEADS) return;
  int e = i >> 2, hd = i & 3;
  int s, d;
  if (e < E) { s = (int)src[e]; d = (int)dst[e]; } else { s = d = e - E; }
  float sc = a_s[s * HEADS + hd] + a_d[d * HEADS + hd];
  sc = sc > 0.f ? sc : 0.2f * sc;                  // leaky relu, slope 0.2
  ebuf[i] = sc;
  atomicMax(nmax + d * HEADS + hd, f2o(sc));
}

// ---- exp(score - max) + segment sum ----
__global__ void k_edge_exp(const long long* __restrict__ dst,
                           float* __restrict__ ebuf, const unsigned* __restrict__ nmax,
                           float* __restrict__ den, int E, int Etot) {
  int i = blockIdx.x * blockDim.x + threadIdx.x;
  if (i >= Etot * HEADS) return;
  int e = i >> 2, hd = i & 3;
  int d = (e < E) ? (int)dst[e] : (e - E);
  float ex = __expf(ebuf[i] - o2f(nmax[d * HEADS + hd]));
  ebuf[i] = ex;
  atomicAdd(den + d * HEADS + hd, ex);
}

// ---- weighted message scatter: out[dst] += h[src] * alpha ----
// 4 edges per 256-thread block; 64 lanes/edge, float4 (b128) row loads.
__global__ __launch_bounds__(256) void k_scatter(
    const long long* __restrict__ src, const long long* __restrict__ dst,
    const float* __restrict__ h, const float* __restrict__ ebuf,
    const float* __restrict__ den, float* __restrict__ out, int E, int Etot) {
  int e = blockIdx.x * 4 + (threadIdx.x >> 6);
  if (e >= Etot) return;
  int t  = threadIdx.x & 63;
  int j4 = t * 4;                  // column base (0..252)
  int hd = j4 >> 6;
  int s, d;
  if (e < E) { s = (int)src[e]; d = (int)dst[e]; } else { s = d = e - E; }
  float alpha = ebuf[e * HEADS + hd] / (den[d * HEADS + hd] + 1e-16f);
  const float4 hv = *(const float4*)(h + (size_t)s * F + j4);
  float* op = out + (size_t)d * F + j4;
  atomicAdd(op + 0, hv.x * alpha);
  atomicAdd(op + 1, hv.y * alpha);
  atomicAdd(op + 2, hv.z * alpha);
  atomicAdd(op + 3, hv.w * alpha);
}

// ---- bias + ELU, emit bf16 operand for next GEMM ----
__global__ void k_elu_cvt(const float* __restrict__ in, const float* __restrict__ bias,
                          __bf16* __restrict__ out, int n) {
  int i = blockIdx.x * blockDim.x + threadIdx.x;
  if (i >= n) return;
  float v = in[i] + bias[i & (F - 1)];
  v = v > 0.f ? v : (__expf(v) - 1.f);
  out[i] = (__bf16)v;
}

// ---- mean over heads + bias2 -> d_out [N,64] ----
__global__ void k_final(const float* __restrict__ acc, const float* __restrict__ bias,
                        float* __restrict__ out, int N) {
  int i = blockIdx.x * blockDim.x + threadIdx.x;
  if (i >= N * CDIM) return;
  int n = i >> 6, c = i & 63;
  const float* p = acc + (size_t)n * F + c;
  out[i] = 0.25f * (p[0] + p[64] + p[128] + p[192]) + bias[c];
}

extern "C" void kernel_launch(void* const* d_in, const int* in_sizes, int n_in,
                              void* d_out, int out_size, void* d_ws, size_t ws_size,
                              hipStream_t stream) {
  const float*     x    = (const float*)d_in[0];
  const long long* ei   = (const long long*)d_in[1];   // int64 [2,E]
  const float*     W1   = (const float*)d_in[2];
  const float*     as1  = (const float*)d_in[3];
  const float*     ad1  = (const float*)d_in[4];
  const float*     b1   = (const float*)d_in[5];
  const float*     W2   = (const float*)d_in[6];
  const float*     as2  = (const float*)d_in[7];
  const float*     ad2  = (const float*)d_in[8];
  const float*     b2   = (const float*)d_in[9];
  float* out = (float*)d_out;

  const int N    = in_sizes[0] / F;      // 50000
  const int E    = in_sizes[1] / 2;      // 400000
  const int Etot = E + N;                // + self loops
  const long long* srcI = ei;
  const long long* dstI = ei + E;

  char* ws = (char*)d_ws;
  size_t off = 0;
  auto alloc = [&](size_t bytes) -> char* {
    char* p = ws + off; off = (off + bytes + 255) & ~(size_t)255; return p;
  };
  float*    bufA = (float*)   alloc((size_t)N * F * 4);   // h1, later g2
  float*    bufB = (float*)   alloc((size_t)N * F * 4);   // out1 accum, later out2 accum
  __bf16*   xbf  = (__bf16*)  alloc((size_t)N * F * 2);
  __bf16*   h2bf = (__bf16*)  alloc((size_t)N * F * 2);
  __bf16*   W1t  = (__bf16*)  alloc((size_t)256 * 256 * 2);
  __bf16*   W2t  = (__bf16*)  alloc((size_t)256 * 256 * 2);
  float*    aS   = (float*)   alloc((size_t)N * HEADS * 4);
  float*    aD   = (float*)   alloc((size_t)N * HEADS * 4);
  unsigned* nmax = (unsigned*)alloc((size_t)N * HEADS * 4);
  float*    den  = (float*)   alloc((size_t)N * HEADS * 4);
  float*    ebuf = (float*)   alloc((size_t)Etot * HEADS * 4);

  const int TB = 256;
  const int nF = N * F;
  const int nH = N * HEADS;
  const int eH = Etot * HEADS;

  // one-time operand prep
  k_cvt_bf16<<<(nF + TB - 1) / TB, TB, 0, stream>>>(x, xbf, nF);
  k_transpose_cvt<<<65536 / TB, TB, 0, stream>>>(W1, W1t);
  k_transpose_cvt<<<65536 / TB, TB, 0, stream>>>(W2, W2t);

  dim3 gg((N + 31) / 32, HEADS);

  // ================= layer 1 =================
  k_gemm_att<<<gg, 32, 0, stream>>>(xbf, W1t, as1, ad1, bufA, aS, aD, N);
  k_init_stats<<<(nH + TB - 1) / TB, TB, 0, stream>>>(nmax, den, nH);
  k_zero<<<(nF / 4 + TB - 1) / TB, TB, 0, stream>>>((float4*)bufB, nF / 4);
  k_edge_max<<<(eH + TB - 1) / TB, TB, 0, stream>>>(srcI, dstI, aS, aD, ebuf, nmax, E, Etot);
  k_edge_exp<<<(eH + TB - 1) / TB, TB, 0, stream>>>(dstI, ebuf, nmax, den, E, Etot);
  k_scatter<<<(Etot + 3) / 4, 256, 0, stream>>>(srcI, dstI, bufA, ebuf, den, bufB, E, Etot);
  k_elu_cvt<<<(nF + TB - 1) / TB, TB, 0, stream>>>(bufB, b1, h2bf, nF);

  // ================= layer 2 =================
  k_gemm_att<<<gg, 32, 0, stream>>>(h2bf, W2t, as2, ad2, bufA, aS, aD, N);
  k_init_stats<<<(nH + TB - 1) / TB, TB, 0, stream>>>(nmax, den, nH);
  k_zero<<<(nF / 4 + TB - 1) / TB, TB, 0, stream>>>((float4*)bufB, nF / 4);
  k_edge_max<<<(eH + TB - 1) / TB, TB, 0, stream>>>(srcI, dstI, aS, aD, ebuf, nmax, E, Etot);
  k_edge_exp<<<(eH + TB - 1) / TB, TB, 0, stream>>>(dstI, ebuf, nmax, den, E, Etot);
  k_scatter<<<(Etot + 3) / 4, 256, 0, stream>>>(srcI, dstI, bufA, ebuf, den, bufB, E, Etot);

  k_final<<<(N * CDIM + TB - 1) / TB, TB, 0, stream>>>(bufB, b2, out, N);
}
